// MultiHeadAttention_1537598292607
// MI455X (gfx1250) — matrix operand order repaired and checked by reading, not compiled, not run on hardware
//
#include <hip/hip_runtime.h>
#include <hip/hip_bf16.h>

typedef __attribute__((ext_vector_type(16))) _Float16 v16h;
typedef __attribute__((ext_vector_type(8)))  _Float16 v8h;
typedef __attribute__((ext_vector_type(4)))  _Float16 v4h;
typedef __attribute__((ext_vector_type(8)))  float    v8f;
typedef __attribute__((ext_vector_type(4)))  unsigned int v4u;
typedef __attribute__((ext_vector_type(8)))  int      v8i;
typedef __attribute__((ext_vector_type(4)))  int      v4i;

#define N_HEADC 16
#define HDIM    64
#define DMODEL  1024
#define BSZ     4
#define SEQ     2048
#define MTOT    (BSZ * SEQ)          // 8192

__device__ __forceinline__ v16h combine16(v8h lo, v8h hi) {
  v16h r;
#pragma unroll
  for (int i = 0; i < 8; ++i) { r[i] = lo[i]; r[i + 8] = hi[i]; }
  return r;
}

// ---------------------------------------------------------------------------
// fp32 -> f16 conversion (vectorized x4)
// ---------------------------------------------------------------------------
__global__ void cvt_f32_f16(const float* __restrict__ src,
                            _Float16* __restrict__ dst, int n4) {
  int i = blockIdx.x * blockDim.x + threadIdx.x;
  if (i >= n4) return;
  float4 v = ((const float4*)src)[i];
  v4h o;
  o[0] = (_Float16)v.x; o[1] = (_Float16)v.y;
  o[2] = (_Float16)v.z; o[3] = (_Float16)v.w;
  ((v4h*)dst)[i] = o;
}

// ---------------------------------------------------------------------------
// WMMA GEMM: C(M x 1024) = A(M x 1024, f16 rm) * W(1024 x 1024, f16 rm)
// Block tile 128x64, 8 waves (4x2), each wave 32x32 (2x2 WMMA subtiles).
// A-tile is staged into LDS by the Tensor Data Mover (TDM) with hardware
// row padding (64B rows + 16B pad -> 40-half LDS stride). Wave 0 issues the
// tensor_load_to_lds and waits TENSORcnt; the block barrier publishes it.
// MODE 0: epilogue applies RoPE, writes f16 (b,h,s,d)
// MODE 1: epilogue writes f16 transposed (b,h,d,s)   (for V)
// MODE 2: epilogue writes fp32 row-major             (final output)
// ---------------------------------------------------------------------------
enum { MODE_ROPE = 0, MODE_VT = 1, MODE_F32 = 2 };

template <int MODE>
__global__ __launch_bounds__(256) void gemm_wmma(
    const _Float16* __restrict__ A, const _Float16* __restrict__ W,
    void* __restrict__ Out) {
  constexpr int K = DMODEL, N = DMODEL;
  constexpr int SA = 40;   // padded LDS strides (multiples of 8 halves)
  constexpr int SB = 40;
  __shared__ __attribute__((aligned(16))) _Float16 ldsA[128 * SA];
  __shared__ __attribute__((aligned(16))) _Float16 ldsB[64 * SB];  // [n][k]

  const int t    = threadIdx.x;
  const int lane = t & 31;
  const int w    = t >> 5;
  const int half = lane >> 4;
  const int l16  = lane & 15;
  const int mb   = blockIdx.y * 128;
  const int nb   = blockIdx.x * 64;
  const int m_off = (w >> 1) * 32;
  const int n_off = (w & 1) * 32;

  // LDS byte offset of ldsA (generic-pointer low 32 bits = LDS offset)
  const unsigned int ldsA_off = (unsigned int)(unsigned long long)(ldsA);

  v8f acc[2][2] = {};

  for (int kb = 0; kb < K; kb += 32) {
    __syncthreads();
    // --- A tile via TDM: 128 rows x 32 halves, row stride 1024, LDS pad ---
    if (w == 0) {
      unsigned long long ga =
          (unsigned long long)(A + (size_t)mb * K + kb);  // byte address
      v4u g0;
      g0[0] = 1u;                                   // count=1 (valid D#)
      g0[1] = ldsA_off;                             // lds_addr
      g0[2] = (unsigned int)ga;                     // global_addr[31:0]
      g0[3] = (unsigned int)((ga >> 32) & 0x1FFFFFFu) | (2u << 30);  // +type=2
      v8i g1;
      g1[0] = (1 << 16)      // data_size = 2 bytes
            | (1 << 20)      // pad_enable
            | (3 << 22)      // pad_interval: 16 DWORDs (64B)
            | (3 << 25);     // pad_amount:   4 DWORDs (16B)
      g1[1] = (int)(32u << 16);    // tensor_dim0[15:0]=32 (bits 79:48)
      g1[2] = (int)(128u << 16);   // tensor_dim0 hi=0 ; tensor_dim1[15:0]=128
      g1[3] = (int)(32u << 16);    // tensor_dim1 hi=0 ; tile_dim0=32
      g1[4] = 128;                 // tile_dim1=128 ; tile_dim2=0
      g1[5] = 1024;                // tensor_dim0_stride[31:0]=1024 elements
      g1[6] = 0;
      g1[7] = 0;
      v4i gz  = {0, 0, 0, 0};
      v8i gz8 = {0, 0, 0, 0, 0, 0, 0, 0};
      __builtin_amdgcn_tensor_load_to_lds(g0, g1, gz, gz, gz8, 0);
      __builtin_amdgcn_s_wait_tensorcnt(0);
    }
    // --- B tile: 32 x 64, stored transposed [n][k] ---
    {
      int krow = t >> 3;
      int c8   = (t & 7) * 8;
      v8h v = *(const v8h*)(W + (size_t)(kb + krow) * N + nb + c8);
#pragma unroll
      for (int j = 0; j < 8; ++j) ldsB[(c8 + j) * SB + krow] = v[j];
      if (kb + 32 < K)  // prefetch next W tile (global_prefetch_b8)
        __builtin_prefetch(W + (size_t)(kb + 32 + krow) * N + nb + c8, 0, 1);
    }
    __syncthreads();

    // --- fragments ---
    v16h af[2], bf[2];
#pragma unroll
    for (int si = 0; si < 2; ++si) {
      const _Float16* base = ldsA + (m_off + si * 16 + l16) * SA;
      v8h lo = *(const v8h*)(base + half * 8);
      v8h hi = *(const v8h*)(base + 16 + half * 8);
      af[si] = combine16(lo, hi);
    }
#pragma unroll
    for (int sj = 0; sj < 2; ++sj) {
      const _Float16* base = ldsB + (n_off + sj * 16 + l16) * SB + half * 8;
      v8h lo = *(const v8h*)(base);
      v8h hi = *(const v8h*)(base + 16);
      bf[sj] = combine16(lo, hi);
    }
#pragma unroll
    for (int si = 0; si < 2; ++si)
#pragma unroll
      for (int sj = 0; sj < 2; ++sj)
        acc[si][sj] = __builtin_amdgcn_wmma_f32_16x16x32_f16(
            false, af[si], false, bf[sj], (short)0, acc[si][sj], false, false);
  }

  // --- epilogue ---
#pragma unroll
  for (int si = 0; si < 2; ++si) {
#pragma unroll
    for (int sj = 0; sj < 2; ++sj) {
      const int ng = nb + n_off + sj * 16 + l16;  // global col in [0,1024)
#pragma unroll
      for (int r = 0; r < 8; ++r) {
        const int mg = mb + m_off + si * 16 + r + 8 * half;  // global row
        float val = acc[si][sj][r];
        if constexpr (MODE == MODE_F32) {
          ((float*)Out)[(size_t)mg * N + ng] = val;
        } else {
          const int b = mg / SEQ, s = mg % SEQ;
          const int hh = ng >> 6, d = ng & 63;
          if constexpr (MODE == MODE_ROPE) {
            // rotate pair (even,odd) lives in adjacent lane
            float part = __shfl_xor(val, 1, 32);
            float invf =
                __expf(-(float)(d & ~1) * (9.210340371976184f / 64.0f));
            float ang = (float)s * invf;
            float sn, cs;
            __sincosf(ang, &sn, &cs);
            float o = (d & 1) ? (part * sn + val * cs)
                              : (val * cs - part * sn);
            ((_Float16*)Out)[(((size_t)b * N_HEADC + hh) * SEQ + s) * HDIM + d] =
                (_Float16)o;
          } else {  // MODE_VT : (b,h,d,s)
            ((_Float16*)Out)[(((size_t)b * N_HEADC + hh) * HDIM + d) * SEQ + s] =
                (_Float16)val;
          }
        }
      }
    }
  }
}

// ---------------------------------------------------------------------------
// One 32-key block of causal flash attention for one wave (16 query rows).
// MASKED=false for blocks fully below the diagonal (no compares emitted).
// Scores arrive pre-scaled: Q fragments carry the 1/sqrt(d) factor.
// ---------------------------------------------------------------------------
template <bool MASKED>
__device__ __forceinline__ void attn_block(
    int kb, int qb, int half, int l16,
    const _Float16* __restrict__ Kbase, const _Float16* __restrict__ Vbase,
    _Float16* ldsW, const v16h (&qf)[2], v8f (&o)[4],
    float (&m_run)[8], float (&s_run)[8]) {
  // ---- scores S = Q K^T : two 16x16 tiles ----
  v8f sc[2] = {};
#pragma unroll
  for (int tt = 0; tt < 2; ++tt) {
#pragma unroll
    for (int c = 0; c < 2; ++c) {
      const _Float16* base =
          Kbase + (size_t)(kb + tt * 16 + l16) * HDIM + c * 32 + half * 8;
      v8h lo = *(const v8h*)(base);
      v8h hi = *(const v8h*)(base + 16);
      v16h kf = combine16(lo, hi);
      sc[tt] = __builtin_amdgcn_wmma_f32_16x16x32_f16(
          false, qf[c], false, kf, (short)0, sc[tt], false, false);
    }
  }
  // ---- online softmax update ----
#pragma unroll
  for (int r = 0; r < 8; ++r) {
    float s0 = sc[0][r];
    float s1 = sc[1][r];
    if constexpr (MASKED) {
      const int mq = qb + r + 8 * half;
      if (kb + l16 > mq) s0 = -1e9f;
      if (kb + 16 + l16 > mq) s1 = -1e9f;
    }
    float rm = fmaxf(s0, s1);
#pragma unroll
    for (int off = 8; off; off >>= 1) rm = fmaxf(rm, __shfl_xor(rm, off, 32));
    float mn  = fmaxf(m_run[r], rm);
    float fac = __expf(m_run[r] - mn);
    float p0  = __expf(s0 - mn);
    float p1  = __expf(s1 - mn);
    float rs  = p0 + p1;
#pragma unroll
    for (int off = 8; off; off >>= 1) rs += __shfl_xor(rs, off, 32);
    s_run[r] = s_run[r] * fac + rs;
    m_run[r] = mn;
#pragma unroll
    for (int c = 0; c < 4; ++c) o[c][r] *= fac;
    const int row = r + 8 * half;
    ldsW[row * 40 + l16]      = (_Float16)p0;
    ldsW[row * 40 + 16 + l16] = (_Float16)p1;
  }
  // ---- P as A-fragment (16x32) from LDS ----
  v16h pf;
  {
    const _Float16* base = ldsW + l16 * 40;
    v8h lo = *(const v8h*)(base + half * 8);
    v8h hi = *(const v8h*)(base + 16 + half * 8);
    pf = combine16(lo, hi);
  }
  // ---- O += P * V ----
#pragma unroll
  for (int c = 0; c < 4; ++c) {
    const _Float16* base =
        Vbase + (size_t)(c * 16 + l16) * SEQ + kb + half * 8;
    v8h lo = *(const v8h*)(base);
    v8h hi = *(const v8h*)(base + 16);
    v16h vf = combine16(lo, hi);
    o[c] = __builtin_amdgcn_wmma_f32_16x16x32_f16(
        false, pf, false, vf, (short)0, o[c], false, false);
  }
}

// ---------------------------------------------------------------------------
// Causal flash attention. Grid: (SEQ/64, B*H). Block: 128 threads (4 waves),
// wave w owns 16 query rows. Online softmax; WMMA for QK^T and P*V.
// Qh, Kh: (b,h,s,64) f16.  Vt: (b,h,64,s) f16.  Oh: (b,s,h*64) f16.
// ---------------------------------------------------------------------------
__global__ __launch_bounds__(128) void attn_wmma(
    const _Float16* __restrict__ Qh, const _Float16* __restrict__ Kh,
    const _Float16* __restrict__ Vt, _Float16* __restrict__ Oh) {
  constexpr int SP = 40;
  __shared__ __attribute__((aligned(16))) _Float16 ldsP[4 * 16 * SP];

  const int t    = threadIdx.x;
  const int lane = t & 31;
  const int w    = t >> 5;
  const int half = lane >> 4;
  const int l16  = lane & 15;
  const int bh   = blockIdx.y;            // b*16 + h
  const int b    = bh >> 4;
  const int h    = bh & 15;
  const int qb   = blockIdx.x * 64 + w * 16;
  const _Float16* Kbase = Kh + (size_t)bh * SEQ * HDIM;
  const _Float16* Vbase = Vt + (size_t)bh * HDIM * SEQ;
  _Float16* ldsW = ldsP + w * 16 * SP;

  // Q fragments (2 k-chunks of 32), pre-scaled by 1/sqrt(64) (exact in f16)
  v16h qf[2];
#pragma unroll
  for (int c = 0; c < 2; ++c) {
    const _Float16* base =
        Qh + (size_t)bh * SEQ * HDIM + (size_t)(qb + l16) * HDIM + c * 32 +
        half * 8;
    v8h lo = *(const v8h*)(base);
    v8h hi = *(const v8h*)(base + 16);
    qf[c] = combine16(lo, hi);
#pragma unroll
    for (int e = 0; e < 16; ++e) qf[c][e] *= (_Float16)0.125f;
  }

  v8f o[4] = {};
  float m_run[8], s_run[8];
#pragma unroll
  for (int r = 0; r < 8; ++r) { m_run[r] = -1e30f; s_run[r] = 0.f; }

  // blocks with kb+31 <= qb need no mask for any of this wave's rows
  const int unmask_end = ((qb + 1) >> 5) << 5;
  for (int kb = 0; kb < unmask_end; kb += 32)
    attn_block<false>(kb, qb, half, l16, Kbase, Vbase, ldsW, qf, o, m_run, s_run);
  for (int kb = unmask_end; kb < qb + 16; kb += 32)
    attn_block<true>(kb, qb, half, l16, Kbase, Vbase, ldsW, qf, o, m_run, s_run);

  // ---- normalize + write (b,s,h*64+d) f16 ----
#pragma unroll
  for (int r = 0; r < 8; ++r) {
    const float inv = 1.0f / s_run[r];
    const int s = qb + r + 8 * half;
    const size_t rowbase = ((size_t)b * SEQ + s) * DMODEL + h * HDIM;
#pragma unroll
    for (int c = 0; c < 4; ++c)
      Oh[rowbase + c * 16 + l16] = (_Float16)(o[c][r] * inv);
  }
}

// ---------------------------------------------------------------------------
extern "C" void kernel_launch(void* const* d_in, const int* in_sizes, int n_in,
                              void* d_out, int out_size, void* d_ws,
                              size_t ws_size, hipStream_t stream) {
  (void)in_sizes; (void)n_in; (void)out_size; (void)ws_size;
  const float* q  = (const float*)d_in[0];
  const float* k  = (const float*)d_in[1];
  const float* v  = (const float*)d_in[2];
  const float* Wq = (const float*)d_in[3];
  const float* Wk = (const float*)d_in[4];
  const float* Wv = (const float*)d_in[5];
  const float* Wo = (const float*)d_in[6];
  // d_in[7] = causal mask, structure known -> unused
  float* out = (float*)d_out;

  const size_t ACT = (size_t)MTOT * DMODEL;    // 8388608
  const size_t WEL = (size_t)DMODEL * DMODEL;  // 1048576

  _Float16* ws = (_Float16*)d_ws;
  _Float16* xq = ws;            // f16 activations
  _Float16* xk = xq + ACT;
  _Float16* xv = xk + ACT;
  _Float16* wq = xv + ACT;      // f16 weights
  _Float16* wk = wq + WEL;
  _Float16* wv = wk + WEL;
  _Float16* wo = wv + WEL;
  _Float16* Qh = wo + WEL;      // (b,h,s,64) rope'd
  _Float16* Khd = Qh + ACT;     // (b,h,s,64) rope'd
  _Float16* Vt = Khd + ACT;     // (b,h,64,s)
  _Float16* Oh = Vt + ACT;      // (b,s,1024)

  const int CB = 256;
  cvt_f32_f16<<<(int)(ACT / 4 + CB - 1) / CB, CB, 0, stream>>>(q, xq, (int)(ACT / 4));
  cvt_f32_f16<<<(int)(ACT / 4 + CB - 1) / CB, CB, 0, stream>>>(k, xk, (int)(ACT / 4));
  cvt_f32_f16<<<(int)(ACT / 4 + CB - 1) / CB, CB, 0, stream>>>(v, xv, (int)(ACT / 4));
  cvt_f32_f16<<<(int)(WEL / 4 + CB - 1) / CB, CB, 0, stream>>>(Wq, wq, (int)(WEL / 4));
  cvt_f32_f16<<<(int)(WEL / 4 + CB - 1) / CB, CB, 0, stream>>>(Wk, wk, (int)(WEL / 4));
  cvt_f32_f16<<<(int)(WEL / 4 + CB - 1) / CB, CB, 0, stream>>>(Wv, wv, (int)(WEL / 4));
  cvt_f32_f16<<<(int)(WEL / 4 + CB - 1) / CB, CB, 0, stream>>>(Wo, wo, (int)(WEL / 4));

  dim3 ggrid(DMODEL / 64, MTOT / 128);  // (16, 64)
  gemm_wmma<MODE_ROPE><<<ggrid, 256, 0, stream>>>(xq, wq, (void*)Qh);
  gemm_wmma<MODE_ROPE><<<ggrid, 256, 0, stream>>>(xk, wk, (void*)Khd);
  gemm_wmma<MODE_VT><<<ggrid, 256, 0, stream>>>(xv, wv, (void*)Vt);

  attn_wmma<<<dim3(SEQ / 64, BSZ * N_HEADC), 128, 0, stream>>>(Qh, Khd, Vt, Oh);

  gemm_wmma<MODE_F32><<<ggrid, 256, 0, stream>>>(Oh, wo, (void*)out);
}
